// SenticGCNClassifier_83614423318853
// MI455X (gfx1250) — matrix operand order repaired and checked by reading, not compiled, not run on hardware
//
#include <hip/hip_runtime.h>
#include <math.h>
#include <stdint.h>

// ---------------------------------------------------------------------------
// Sentic-GCN classifier for MI455X (gfx1250), wave32 + WMMA f32_16x16x32_f16.
// Round 4: round-3 disasm showed scratch spills in the LSTM loop (4-acc
// scheme + 1024 threads starved the register allocator). Fix: 512-thread
// LSTM block (16 waves -> 2x VGPR budget/wave) and a rolled k-loop
// (#pragma unroll 1) so A fragments are not hoisted+spilled. Keeps the
// 1-B-load -> 4-independent-WMMA structure.
// ---------------------------------------------------------------------------

typedef __attribute__((ext_vector_type(16))) _Float16 v16h;
typedef __attribute__((ext_vector_type(8)))  float    v8f;

#define B_    64
#define L_    256
#define E_    300
#define HID_  300
#define DM_   600      // 2*HID
#define DMP_  608      // DM padded to mult of 32 (19*32)
#define G4_   1200     // 4*HID (gate width) -- 75*16, mult of 16
#define KIN_  608      // LSTM input dim 600 padded to 608
#define KH_   320      // hidden 300 padded to 320 (10*32)
#define AL_   4
#define P_    3
#define NROW_ (B_*L_)  // 16384

#define LSTM_T_ 512    // LSTM block size (16 waves: high VGPR budget, no spill)
#define CREG_   38     // ceil(64*300 / 512) c-state registers per thread

// ---------------------------------------------------------------------------
// WMMA fragment loaders (layouts per CDNA5 ISA 7.12.2, 16-bit A 16x32 and
// B 32x16). Rows of A / rows of W(N,K) are contiguous in K, so each lane
// loads two aligned 16B chunks.
// ---------------------------------------------------------------------------
__device__ inline v16h load_a_frag(const _Float16* __restrict__ A, int lda,
                                   int row0, int k0, int lane) {
  const int m    = lane & 15;
  const int ksel = (lane >> 4) ? 8 : 0;   // lanes 16-31 hold K {8..15,24..31}
  const _Float16* p = A + (size_t)(row0 + m) * lda + k0 + ksel;
  v16h r;
  *((float4*)&r)     = *(const float4*)(p);        // K +0..7
  *((float4*)&r + 1) = *(const float4*)(p + 16);   // K +16..23
  return r;
}

__device__ inline v16h load_b_frag(const _Float16* __restrict__ W, int ldw,
                                   int n0, int k0, int lane) {
  const int n    = lane & 15;
  const int koff = (lane >> 4) ? 8 : 0;   // lanes 16-31 hold K {8..15,24..31}
  const _Float16* p = W + (size_t)(n0 + n) * ldw + k0 + koff;
  v16h r;
  *((float4*)&r)     = *(const float4*)(p);
  *((float4*)&r + 1) = *(const float4*)(p + 16);
  return r;
}

// ---------------------------------------------------------------------------
// Generic WMMA GEMM: C[M,Npad] = A[M,K](f16,row-major) * W[Npad,K](f16)^T
// (+bias over real N) (+optional relu). One wave computes a 16x64 strip,
// block = 8 waves covers 16x512, grid.z = batch. One A load feeds 4 WMMAs.
// ---------------------------------------------------------------------------
template <bool HALF_OUT, bool RELU>
__global__ __launch_bounds__(256)
void wmma_gemm_kernel(const _Float16* __restrict__ A, size_t aBatch, int lda,
                      const _Float16* __restrict__ W, size_t wBatch, int ldw,
                      void* __restrict__ Cv, size_t cBatch, int ldc,
                      int M, int Nreal, int Npad, int K,
                      const float* __restrict__ bias) {
  const int lane = threadIdx.x & 31;
  const int wave = threadIdx.x >> 5;
  const int bz   = blockIdx.z;
  A += (size_t)bz * aBatch;
  W += (size_t)bz * wBatch;

  const int row0 = blockIdx.y * 16;
  if (row0 >= M) return;                       // block-uniform
  const int tn0 = blockIdx.x * 512 + wave * 64;
  if (tn0 >= Npad) return;                     // wave-uniform
  int nsub = (Npad - tn0) >> 4; if (nsub > 4) nsub = 4;  // wave-uniform

  const v8f vz = {0.f,0.f,0.f,0.f,0.f,0.f,0.f,0.f};
  v8f acc[4] = {vz, vz, vz, vz};

  for (int k0 = 0; k0 < K; k0 += 32) {
    v16h a = load_a_frag(A, lda, row0, k0, lane);
#pragma unroll
    for (int s = 0; s < 4; ++s) {
      if (s < nsub) {
        v16h b = load_b_frag(W, ldw, tn0 + s * 16, k0, lane);
        acc[s] = __builtin_amdgcn_wmma_f32_16x16x32_f16(
            false, a, false, b, (short)0, acc[s], false, false);
      }
    }
  }

  const int nn = lane & 15;
  const int mb = (lane >> 4) ? 8 : 0;
#pragma unroll
  for (int s = 0; s < 4; ++s) {
    if (s < nsub) {
      const int col = tn0 + s * 16 + nn;
      float bv = 0.f;
      if (bias != nullptr && col < Nreal) bv = bias[col];
#pragma unroll
      for (int r = 0; r < 8; ++r) {
        float v = acc[s][r] + bv;
        if (RELU) v = fmaxf(v, 0.f);
        const size_t off = (size_t)bz * cBatch +
                           (size_t)(row0 + mb + r) * ldc + col;
        if (HALF_OUT) ((_Float16*)Cv)[off] = (_Float16)v;
        else          ((float*)Cv)[off]    = v;
      }
    }
  }
}

// ---------------------------------------------------------------------------
// Small helper kernels
// ---------------------------------------------------------------------------
__global__ void addvec_kernel(const float* a, const float* b, float* o, int n) {
  const int i = blockIdx.x * blockDim.x + threadIdx.x;
  if (i < n) o[i] = a[i] + b[i];
}

// float [rows,cols] -> f16 [rowsPad,colsPad], zero padded
__global__ void convpad_kernel(const float* __restrict__ src,
                               _Float16* __restrict__ dst,
                               int rows, int cols, int colsPad) {
  const int r = blockIdx.x;
  for (int c = threadIdx.x; c < colsPad; c += blockDim.x) {
    float v = (r < rows && c < cols) ? src[(size_t)r * cols + c] : 0.f;
    dst[(size_t)r * colsPad + c] = (_Float16)v;
  }
}

__global__ void aspavg_kernel(const int* __restrict__ asp,
                              const float* __restrict__ emb_table,
                              float* __restrict__ asp_avg) {
  const int b = blockIdx.x, e = threadIdx.x;
  if (e < E_) {
    float s = 0.f;
    for (int k = 0; k < AL_; ++k)
      s += emb_table[(size_t)asp[b * AL_ + k] * E_ + e];
    asp_avg[b * E_ + e] = s * (1.f / AL_);
  }
}

// emb16[b*L+t] = concat(word_emb, asp_avg[b]) padded to KIN_, f16
__global__ void embed_kernel(const int* __restrict__ tok,
                             const float* __restrict__ emb_table,
                             const float* __restrict__ asp_avg,
                             _Float16* __restrict__ emb16) {
  const int row = blockIdx.x;            // b*L + t
  const int b   = row / L_;
  const float* we = emb_table + (size_t)tok[row] * E_;
  for (int col = threadIdx.x; col < KIN_; col += blockDim.x) {
    float v = 0.f;
    if (col < E_)          v = we[col];
    else if (col < 2 * E_) v = asp_avg[b * E_ + (col - E_)];
    emb16[(size_t)row * KIN_ + col] = (_Float16)v;
  }
}

__global__ void amask_kernel(const int* __restrict__ asp,
                             int* __restrict__ amask) {
  const int b = blockIdx.x, t = threadIdx.x;
  int m = 0;
  for (int k = 0; k < AL_; ++k) m |= (asp[b * AL_ + k] == t) ? 1 : 0;
  amask[b * L_ + t] = m;
}

// Closed-form adjacency A[b][i][j] (no scatter: each column j of M has one
// entry at row head[j]-1 when head[j]>0, so D = OR of eye, M, M^T).
__device__ inline float adj_val(int b, int i, int j,
                                const int* __restrict__ tok,
                                const int* __restrict__ head,
                                const float* __restrict__ sentic,
                                const int* __restrict__ amask) {
  const float pi = sentic[tok[b * L_ + i]];
  const float pj = sentic[tok[b * L_ + j]];
  const float nz = (pi != 0.f && pj != 0.f) ? 1.f : 0.f;
  const float S  = (pi + pj) * nz;
  const float T  = (amask[b * L_ + i] | amask[b * L_ + j]) ? 1.f : 0.f;
  const int hi = head[b * L_ + i], hj = head[b * L_ + j];
  const bool d = (i == j) || (hj > 0 && (hj - 1) == i) || (hi > 0 && (hi - 1) == j);
  return d ? (S + T + 1.f) : 0.f;
}

__global__ void rowsum_kernel(const int* tok, const int* head,
                              const float* sentic, const int* amask,
                              float* __restrict__ dinv) {
  const int b = blockIdx.y, i = blockIdx.x, j = threadIdx.x;
  __shared__ float red[L_];
  red[j] = adj_val(b, i, j, tok, head, sentic, amask);
  __syncthreads();
  for (int o = L_ / 2; o > 0; o >>= 1) {
    if (j < o) red[j] += red[j + o];
    __syncthreads();
  }
  if (j == 0) dinv[b * L_ + i] = rsqrtf(red[0] + 1e-8f);
}

__global__ void ahat_kernel(const int* tok, const int* head,
                            const float* sentic, const int* amask,
                            const float* __restrict__ dinv,
                            _Float16* __restrict__ Ahat) {
  const int b = blockIdx.y, i = blockIdx.x, j = threadIdx.x;
  const float v = adj_val(b, i, j, tok, head, sentic, amask) *
                  dinv[b * L_ + i] * dinv[b * L_ + j];
  Ahat[((size_t)b * L_ + i) * L_ + j] = (_Float16)v;
}

// support f32 [B*L, DMP_] -> supT f16 [B, DMP_, L_]
__global__ void transpose_kernel(const float* __restrict__ support,
                                 _Float16* __restrict__ supT) {
  const int b = blockIdx.y, d = blockIdx.x, t = threadIdx.x;
  supT[((size_t)b * DMP_ + d) * L_ + t] =
      (_Float16)support[((size_t)(b * L_ + t)) * DMP_ + d];
}

// ---------------------------------------------------------------------------
// Bi-LSTM recurrence: one persistent workgroup per direction, 16 waves.
// h-state (f16, padded K) lives in LDS; c-state lives in registers;
// gate pre-activations round-trip through an L2-resident global buffer.
// One wave owns a full 64-row x 16-col strip: per k-step 1 B load -> 4 WMMAs.
// ---------------------------------------------------------------------------
__device__ inline float sigf(float x) { return 1.f / (1.f + expf(-x)); }

__global__ __launch_bounds__(LSTM_T_)
void lstm_kernel(const _Float16* __restrict__ Gin_f,
                 const _Float16* __restrict__ Gin_b,
                 const _Float16* __restrict__ Whh_f,
                 const _Float16* __restrict__ Whh_b,
                 float* __restrict__ gbuf,        // [2][B_*G4_] f32
                 _Float16* __restrict__ Hs) {     // [B*L][DMP_] f16
  __shared__ __align__(16) _Float16 h16[B_ * KH_];
  const int dir = blockIdx.x;
  const _Float16* __restrict__ Gin = dir ? Gin_b : Gin_f;
  const _Float16* __restrict__ Whh = dir ? Whh_b : Whh_f;
  float* __restrict__ g = gbuf + (size_t)dir * (B_ * G4_);

  const int tid  = threadIdx.x;
  const int lane = tid & 31;
  const int wave = tid >> 5;          // 0..15

  for (int i = tid; i < B_ * KH_; i += LSTM_T_) h16[i] = (_Float16)0.f;
  float creg[CREG_];
#pragma unroll
  for (int i = 0; i < CREG_; ++i) creg[i] = 0.f;
  __syncthreads();

  const v8f vz = {0.f,0.f,0.f,0.f,0.f,0.f,0.f,0.f};

  for (int ts = 0; ts < L_; ++ts) {
    const int t = dir ? (L_ - 1 - ts) : ts;

    // ---- phase A: g = h @ Whh^T + Gin[:, t, :]  (M=64, N=1200, K=320)
    // 75 N-strips of 16 over 16 waves; each wave does all 4 M-tiles of its
    // strip so each B fragment load feeds 4 independent WMMAs. k-loop kept
    // rolled so the live set stays bounded (no scratch spills).
    for (int strip = wave; strip < 75; strip += 16) {
      const int tn = strip * 16;
      v8f acc[4] = {vz, vz, vz, vz};
#pragma unroll 1
      for (int k0 = 0; k0 < KH_; k0 += 32) {
        v16h b  = load_b_frag(Whh, KH_, tn, k0, lane);   // global (L2)
        v16h a0 = load_a_frag(h16, KH_,  0, k0, lane);   // LDS
        v16h a1 = load_a_frag(h16, KH_, 16, k0, lane);
        v16h a2 = load_a_frag(h16, KH_, 32, k0, lane);
        v16h a3 = load_a_frag(h16, KH_, 48, k0, lane);
        acc[0] = __builtin_amdgcn_wmma_f32_16x16x32_f16(
            false, a0, false, b, (short)0, acc[0], false, false);
        acc[1] = __builtin_amdgcn_wmma_f32_16x16x32_f16(
            false, a1, false, b, (short)0, acc[1], false, false);
        acc[2] = __builtin_amdgcn_wmma_f32_16x16x32_f16(
            false, a2, false, b, (short)0, acc[2], false, false);
        acc[3] = __builtin_amdgcn_wmma_f32_16x16x32_f16(
            false, a3, false, b, (short)0, acc[3], false, false);
      }
      const int nn  = lane & 15;
      const int mb  = (lane >> 4) ? 8 : 0;
      const int col = tn + nn;
#pragma unroll
      for (int q = 0; q < 4; ++q) {
#pragma unroll
        for (int r = 0; r < 8; ++r) {
          const int m = q * 16 + mb + r;  // batch index
          g[m * G4_ + col] =
              acc[q][r] + (float)Gin[((size_t)m * L_ + t) * G4_ + col];
        }
      }
    }

    // ---- prefetch next timestep's Gin rows (64 x 2400B) into caches
    if (ts + 1 < L_) {
      const int tn2 = dir ? (t - 1) : (t + 1);
      for (int idx = tid; idx < 1200; idx += LSTM_T_) {
        const int m = idx / 19;          // batch row
        const int r = idx - m * 19;      // 128B chunk within the row
        __builtin_prefetch(
            (const void*)(Gin + ((size_t)m * L_ + tn2) * G4_ + r * 64), 0, 3);
      }
    }
    __syncthreads();

    // ---- phase B: gates, state update, write h
#pragma unroll
    for (int it = 0; it < CREG_; ++it) {
      const int idx = tid + it * LSTM_T_;
      if (idx < B_ * HID_) {
        const int b = idx / HID_;
        const int j = idx - b * HID_;
        const float gi = g[b * G4_ + j];
        const float gf = g[b * G4_ + HID_ + j];
        const float gg = g[b * G4_ + 2 * HID_ + j];
        const float go = g[b * G4_ + 3 * HID_ + j];
        const float c  = sigf(gf) * creg[it] + sigf(gi) * tanhf(gg);
        creg[it] = c;
        const float h = sigf(go) * tanhf(c);
        h16[b * KH_ + j] = (_Float16)h;
        Hs[((size_t)b * L_ + t) * DMP_ + dir * HID_ + j] = (_Float16)h;
      }
    }
    __syncthreads();
  }
}

// ---------------------------------------------------------------------------
// Attention softmax + pooling + classifier, one block per batch sample.
// ---------------------------------------------------------------------------
__global__ __launch_bounds__(L_)
void finalize_kernel(const float* __restrict__ gcn,     // [B*L][DMP_]
                     const int* __restrict__ amask,     // [B*L]
                     const float* __restrict__ attn_w,  // [DM_]
                     const float* __restrict__ attn_b,  // [1]
                     const float* __restrict__ cls_W,   // [P_][DM_]
                     const float* __restrict__ cls_b,   // [P_]
                     float* __restrict__ out) {         // [B][P_]
  __shared__ float wsh[L_];
  __shared__ float pooled[DM_];
  __shared__ float red[L_];
  const int b = blockIdx.x, t = threadIdx.x;
  const float* gb = gcn + (size_t)b * L_ * DMP_;

  float s = attn_b[0];
  for (int d = 0; d < DM_; ++d) s += gb[(size_t)t * DMP_ + d] * attn_w[d];
  if (!amask[b * L_ + t]) s = -INFINITY;

  red[t] = s; __syncthreads();
  for (int o = L_ / 2; o > 0; o >>= 1) {
    if (t < o) red[t] = fmaxf(red[t], red[t + o]);
    __syncthreads();
  }
  const float mx = red[0]; __syncthreads();
  const float e = expf(s - mx);
  red[t] = e; __syncthreads();
  for (int o = L_ / 2; o > 0; o >>= 1) {
    if (t < o) red[t] += red[t + o];
    __syncthreads();
  }
  wsh[t] = e / red[0];
  __syncthreads();

  for (int d = t; d < DM_; d += L_) {
    float acc = 0.f;
    for (int tt = 0; tt < L_; ++tt) acc += wsh[tt] * gb[(size_t)tt * DMP_ + d];
    pooled[d] = acc;
  }
  __syncthreads();

  if (t < P_) {
    float acc = cls_b[t];
    for (int d = 0; d < DM_; ++d) acc += pooled[d] * cls_W[t * DM_ + d];
    out[b * P_ + t] = acc;
  }
}

// ---------------------------------------------------------------------------
// Host orchestration
// ---------------------------------------------------------------------------
static inline size_t alignUp256(size_t x) { return (x + 255) & ~(size_t)255; }

extern "C" void kernel_launch(void* const* d_in, const int* in_sizes, int n_in,
                              void* d_out, int out_size, void* d_ws,
                              size_t ws_size, hipStream_t stream) {
  (void)in_sizes; (void)n_in; (void)out_size; (void)ws_size;

  const int*   tok        = (const int*)d_in[0];
  const int*   asp        = (const int*)d_in[1];
  const int*   head       = (const int*)d_in[2];
  const float* emb_table  = (const float*)d_in[3];
  const float* sentic     = (const float*)d_in[4];
  const float* Wih_f      = (const float*)d_in[5];
  const float* Whh_f      = (const float*)d_in[6];
  const float* bih_f      = (const float*)d_in[7];
  const float* bhh_f      = (const float*)d_in[8];
  const float* Wih_b      = (const float*)d_in[9];
  const float* Whh_b      = (const float*)d_in[10];
  const float* bih_b      = (const float*)d_in[11];
  const float* bhh_b      = (const float*)d_in[12];
  const float* gcn_W      = (const float*)d_in[13];
  const float* gcn_b      = (const float*)d_in[14];
  const float* attn_w     = (const float*)d_in[15];
  const float* attn_b     = (const float*)d_in[16];
  const float* cls_W      = (const float*)d_in[17];
  const float* cls_b      = (const float*)d_in[18];
  float*       out        = (float*)d_out;

  // ---- workspace carve (~230 MB)
  uint8_t* p = (uint8_t*)d_ws;
  auto carve = [&](size_t bytes) -> void* {
    void* r = (void*)p; p += alignUp256(bytes); return r;
  };
  float*     asp_avg  = (float*)    carve((size_t)B_ * E_ * 4);
  _Float16*  emb16    = (_Float16*) carve((size_t)NROW_ * KIN_ * 2);
  _Float16*  Wih16f   = (_Float16*) carve((size_t)G4_ * KIN_ * 2);
  _Float16*  Wih16b   = (_Float16*) carve((size_t)G4_ * KIN_ * 2);
  _Float16*  Whh16f   = (_Float16*) carve((size_t)G4_ * KH_ * 2);
  _Float16*  Whh16b   = (_Float16*) carve((size_t)G4_ * KH_ * 2);
  float*     biasf    = (float*)    carve((size_t)G4_ * 4);
  float*     biasb    = (float*)    carve((size_t)G4_ * 4);
  _Float16*  Gin16f   = (_Float16*) carve((size_t)NROW_ * G4_ * 2);
  _Float16*  Gin16b   = (_Float16*) carve((size_t)NROW_ * G4_ * 2);
  float*     gbuf     = (float*)    carve((size_t)2 * B_ * G4_ * 4);
  _Float16*  Hs16     = (_Float16*) carve((size_t)NROW_ * DMP_ * 2);
  _Float16*  gcnW16   = (_Float16*) carve((size_t)DMP_ * DMP_ * 2);
  float*     supportf = (float*)    carve((size_t)NROW_ * DMP_ * 4);
  _Float16*  supT16   = (_Float16*) carve((size_t)B_ * DMP_ * L_ * 2);
  int*       amask    = (int*)      carve((size_t)NROW_ * 4);
  float*     dinv     = (float*)    carve((size_t)NROW_ * 4);
  _Float16*  Ahat16   = (_Float16*) carve((size_t)B_ * L_ * L_ * 2);
  float*     gcn_out  = (float*)    carve((size_t)NROW_ * DMP_ * 4);

  // ---- stage 0: biases, f16 weight conversion, embeddings, masks
  addvec_kernel<<<(G4_ + 255) / 256, 256, 0, stream>>>(bih_f, bhh_f, biasf, G4_);
  addvec_kernel<<<(G4_ + 255) / 256, 256, 0, stream>>>(bih_b, bhh_b, biasb, G4_);
  convpad_kernel<<<G4_, 256, 0, stream>>>(Wih_f, Wih16f, G4_, 2 * E_, KIN_);
  convpad_kernel<<<G4_, 256, 0, stream>>>(Wih_b, Wih16b, G4_, 2 * E_, KIN_);
  convpad_kernel<<<G4_, 256, 0, stream>>>(Whh_f, Whh16f, G4_, HID_, KH_);
  convpad_kernel<<<G4_, 256, 0, stream>>>(Whh_b, Whh16b, G4_, HID_, KH_);
  convpad_kernel<<<DMP_, 256, 0, stream>>>(gcn_W, gcnW16, DM_, DM_, DMP_);
  aspavg_kernel<<<B_, 320, 0, stream>>>(asp, emb_table, asp_avg);
  embed_kernel<<<NROW_, 256, 0, stream>>>(tok, emb_table, asp_avg, emb16);
  amask_kernel<<<B_, L_, 0, stream>>>(asp, amask);
  hipMemsetAsync(Hs16, 0, (size_t)NROW_ * DMP_ * 2, stream);

  // ---- stage 1: LSTM input projections (WMMA GEMM, M=16384 N=1200 K=608)
  {
    dim3 grid((G4_ + 511) / 512, NROW_ / 16, 1);
    wmma_gemm_kernel<true, false><<<grid, 256, 0, stream>>>(
        emb16, 0, KIN_, Wih16f, 0, KIN_, (void*)Gin16f, 0, G4_,
        NROW_, G4_, G4_, KIN_, biasf);
    wmma_gemm_kernel<true, false><<<grid, 256, 0, stream>>>(
        emb16, 0, KIN_, Wih16b, 0, KIN_, (void*)Gin16b, 0, G4_,
        NROW_, G4_, G4_, KIN_, biasb);
  }

  // ---- stage 2: recurrence (one persistent block per direction)
  lstm_kernel<<<2, LSTM_T_, 0, stream>>>(Gin16f, Gin16b, Whh16f, Whh16b,
                                         gbuf, Hs16);

  // ---- stage 3: normalized adjacency A_hat (f16)
  {
    dim3 grid(L_, B_, 1);
    rowsum_kernel<<<grid, L_, 0, stream>>>(tok, head, sentic, amask, dinv);
    ahat_kernel<<<grid, L_, 0, stream>>>(tok, head, sentic, amask, dinv, Ahat16);
  }

  // ---- stage 4: support = Hs @ gcn_W^T + b (M=16384 N=600(608) K=608)
  {
    dim3 grid((DMP_ + 511) / 512, NROW_ / 16, 1);
    wmma_gemm_kernel<false, false><<<grid, 256, 0, stream>>>(
        Hs16, 0, DMP_, gcnW16, 0, DMP_, (void*)supportf, 0, DMP_,
        NROW_, DM_, DMP_, DMP_, gcn_b);
  }

  // ---- stage 5: transpose support -> [B, DMP_, L_] f16 (B operand)
  {
    dim3 grid(DMP_, B_, 1);
    transpose_kernel<<<grid, L_, 0, stream>>>(supportf, supT16);
  }

  // ---- stage 6: gcn_out = relu(A_hat @ support), batched (M=256 N=600 K=256)
  {
    dim3 grid((DMP_ + 511) / 512, L_ / 16, B_);
    wmma_gemm_kernel<false, true><<<grid, 256, 0, stream>>>(
        Ahat16, (size_t)L_ * L_, L_,
        supT16, (size_t)DMP_ * L_, L_,
        (void*)gcn_out, (size_t)L_ * DMP_, DMP_,
        L_, DM_, DMP_, L_, nullptr);
  }

  // ---- stage 7: attention softmax + pooling + classifier
  finalize_kernel<<<B_, L_, 0, stream>>>(gcn_out, amask, attn_w, attn_b,
                                         cls_W, cls_b, out);
}
